// HeteroGNN_18141941858521
// MI455X (gfx1250) — matrix operand order repaired and checked by reading, not compiled, not run on hardware
//
#include <hip/hip_runtime.h>
#include <hip/hip_bf16.h>
#include <stdint.h>

#define N_SRC   100000
#define N_TGT   100000
#define N_EDGES 1600000
#define D       128
#define HID     128
#define OUTD    64

typedef __attribute__((ext_vector_type(16))) __bf16 v16bf;
typedef __attribute__((ext_vector_type(8)))  __bf16 v8bf;
typedef __attribute__((ext_vector_type(4)))  __bf16 v4bf;
typedef __attribute__((ext_vector_type(8)))  float  v8f;

union AFrag { v16bf v; v8bf h[2]; };

__device__ __forceinline__ uint32_t hash_u32(uint32_t x) {
  x ^= x >> 16; x *= 0x7feb352dU; x ^= x >> 15; x *= 0x846ca68bU; x ^= x >> 16;
  return x;
}

__device__ __forceinline__ void atomAddF(float* p, float v) {
  __hip_atomic_fetch_add(p, v, __ATOMIC_RELAXED, __HIP_MEMORY_SCOPE_AGENT);
}

// ---------------------------------------------------------------- utilities
__global__ void zero_f4_kernel(float4* p, int n4) {
  int i = blockIdx.x * blockDim.x + threadIdx.x;
  if (i < n4) p[i] = make_float4(0.f, 0.f, 0.f, 0.f);
}

// Pack [Wl;Wr] (256 x HID, f32) into per-lane WMMA B-fragment order (bf16):
// linear idx = ((ntile*8 + kstep)*32 + lane)*16 + j
__global__ void pack_wcat_kernel(const float* __restrict__ Wl, const float* __restrict__ Wr,
                                 __bf16* __restrict__ out) {
  int idx = blockIdx.x * blockDim.x + threadIdx.x;
  if (idx >= 8 * 8 * 32 * 16) return;
  const int j     = idx & 15;
  const int lane  = (idx >> 4) & 31;
  const int ks    = (idx >> 9) & 7;
  const int ntile = idx >> 12;
  const int k = ks * 32 + (lane >> 4) * 16 + j;   // 0..255
  const int n = ntile * 16 + (lane & 15);         // 0..127
  const float v = (k < D) ? Wl[k * HID + n] : Wr[(k - D) * HID + n];
  out[idx] = (__bf16)v;
}

// Pack W_out (HID x OUTD, f32) the same way: idx = ((nt*4 + ks)*32 + lane)*16 + j
__global__ void pack_wout_kernel(const float* __restrict__ W, __bf16* __restrict__ out) {
  int idx = blockIdx.x * blockDim.x + threadIdx.x;
  if (idx >= 4 * 4 * 32 * 16) return;
  const int j    = idx & 15;
  const int lane = (idx >> 4) & 31;
  const int ks   = (idx >> 9) & 3;
  const int nt   = idx >> 11;
  const int k = ks * 32 + (lane >> 4) * 16 + j;   // 0..127
  const int n = nt * 16 + (lane & 15);            // 0..63
  out[idx] = (__bf16)W[k * OUTD + n];
}

// ------------------------------------------------------- edge scatter (mean agg)
// one wave32 per edge: 4 floats per lane => 128-wide row gather + atomic scatter
template <bool SRC_BF16>
__global__ __launch_bounds__(256)
void scatter_kernel(const void* __restrict__ xsrc_, const int* __restrict__ src_idx,
                    const int* __restrict__ dst_idx, float* __restrict__ agg,
                    float* __restrict__ cnt, int n_edges) {
  const int lane = threadIdx.x & 31;
  const int e = blockIdx.x * 8 + (threadIdx.x >> 5);
  if (e >= n_edges) return;
  const int s = src_idx[e];
  const int d = dst_idx[e];
  float v0, v1, v2, v3;
  if (SRC_BF16) {
    v4bf b = *reinterpret_cast<const v4bf*>((const __bf16*)xsrc_ + (size_t)s * D + lane * 4);
    v0 = (float)b[0]; v1 = (float)b[1]; v2 = (float)b[2]; v3 = (float)b[3];
  } else {
    float4 f = *reinterpret_cast<const float4*>((const float*)xsrc_ + (size_t)s * D + lane * 4);
    v0 = f.x; v1 = f.y; v2 = f.z; v3 = f.w;
  }
  float* ap = agg + (size_t)d * D + lane * 4;
  atomAddF(ap + 0, v0);
  atomAddF(ap + 1, v1);
  atomAddF(ap + 2, v2);
  atomAddF(ap + 3, v3);
  if (lane == 0) atomAddF(cnt + d, 1.0f);
}

// ------------------------------------------------- fused SAGE layer (WMMA bf16)
// 32 rows x 128 cols per block; each wave owns one 16-col tile with two
// 16x16 accumulators. B fragments come pre-packed (1 contiguous 32B load/lane).
// Epilogue: bias + leaky-relu + dropout. FUSE_OUT: project through Wout+bout.
template <bool XDST_BF16, bool FUSE_OUT>
__global__ __launch_bounds__(256)
void sage_gemm_kernel(const float* __restrict__ agg, const float* __restrict__ cnt,
                      const void* __restrict__ xdst_,
                      const __bf16* __restrict__ Bpack,     // [8][8][32][16] bf16
                      const float* __restrict__ bias,       // [HID]
                      __bf16* __restrict__ xout,            // [M][HID] (if !FUSE_OUT)
                      const __bf16* __restrict__ WoutPack,  // [4][4][32][16] (if FUSE_OUT)
                      const float* __restrict__ bout,       // [OUTD]
                      float* __restrict__ fout,             // [M][OUTD]
                      uint32_t drop_tag) {
  __shared__ __align__(16) __bf16 ldsA[32 * 256];  // 16 KB
  __shared__ float ldsInv[32];

  const int tid  = threadIdx.x;
  const int lane = tid & 31;
  const int wave = tid >> 5;
  const int m0   = blockIdx.x * 32;

  if (tid < 32) {
    float c = cnt[m0 + tid];
    ldsInv[tid] = 1.0f / fmaxf(c, 1.0f);
  }
  __syncthreads();

  // Stage A = [mean | x_dst] as bf16: 32 rows x 256 cols, 4 elems/thread/iter
  for (int it = 0; it < 8; ++it) {
    const int eidx = it * 1024 + tid * 4;
    const int r = eidx >> 8;
    const int k = eidx & 255;
    float4 v;
    if (k < D) {
      v = *reinterpret_cast<const float4*>(agg + (size_t)(m0 + r) * D + k);
      const float s = ldsInv[r];
      v.x *= s; v.y *= s; v.z *= s; v.w *= s;
    } else {
      const int kk = k - D;
      if (XDST_BF16) {
        v4bf b = *reinterpret_cast<const v4bf*>((const __bf16*)xdst_ + (size_t)(m0 + r) * D + kk);
        v = make_float4((float)b[0], (float)b[1], (float)b[2], (float)b[3]);
      } else {
        v = *reinterpret_cast<const float4*>((const float*)xdst_ + (size_t)(m0 + r) * D + kk);
      }
    }
    v4bf o;
    o[0] = (__bf16)v.x; o[1] = (__bf16)v.y; o[2] = (__bf16)v.z; o[3] = (__bf16)v.w;
    *reinterpret_cast<v4bf*>(&ldsA[r * 256 + k]) = o;
  }
  __syncthreads();

  const int arow = lane & 15;
  const int asel = lane >> 4;               // K-half select: lane half h holds k 16h..16h+15 (contiguous)
  const int bcol = wave * 16 + arow;        // this wave's 16-col tile of HID

  v8f acc0 = {}, acc1 = {};
  const __bf16* bp = Bpack + ((size_t)wave * 8 * 32 + lane) * 16;
  for (int ks = 0; ks < 8; ++ks) {
    const v16bf b = *reinterpret_cast<const v16bf*>(bp + (size_t)ks * 32 * 16);
    const int kb = ks * 32 + asel * 16;
    AFrag a0, a1;
    a0.h[0] = *reinterpret_cast<const v8bf*>(&ldsA[arow * 256 + kb]);
    a0.h[1] = *reinterpret_cast<const v8bf*>(&ldsA[arow * 256 + kb + 8]);
    a1.h[0] = *reinterpret_cast<const v8bf*>(&ldsA[(arow + 16) * 256 + kb]);
    a1.h[1] = *reinterpret_cast<const v8bf*>(&ldsA[(arow + 16) * 256 + kb + 8]);
    acc0 = __builtin_amdgcn_wmma_f32_16x16x32_bf16(false, a0.v, false, b, (short)0, acc0, false, false);
    acc1 = __builtin_amdgcn_wmma_f32_16x16x32_bf16(false, a1.v, false, b, (short)0, acc1, false, false);
  }

  // epilogue: bias + leaky relu + dropout (both M tiles)
  const uint32_t cutoff = 429496730u;  // P_DROP * 2^32
  const float inv_keep = 1.0f / 0.9f;
  const float bval = bias[bcol];
  float res0[8], res1[8];
#pragma unroll
  for (int r = 0; r < 8; ++r) {
    {
      const int row_g = m0 + r + asel * 8;
      float v = acc0[r] + bval;
      v = (v > 0.f) ? v : 0.01f * v;
      const uint32_t h = hash_u32(((uint32_t)(row_g * HID + bcol)) ^ (drop_tag * 0x9E3779B9u));
      res0[r] = (h >= cutoff) ? v * inv_keep : 0.f;
    }
    {
      const int row_g = m0 + 16 + r + asel * 8;
      float v = acc1[r] + bval;
      v = (v > 0.f) ? v : 0.01f * v;
      const uint32_t h = hash_u32(((uint32_t)(row_g * HID + bcol)) ^ (drop_tag * 0x9E3779B9u));
      res1[r] = (h >= cutoff) ? v * inv_keep : 0.f;
    }
  }

  if (!FUSE_OUT) {
#pragma unroll
    for (int r = 0; r < 8; ++r) {
      const int row_l = r + asel * 8;
      xout[(size_t)(m0 + row_l) * HID + bcol]      = (__bf16)res0[r];
      xout[(size_t)(m0 + 16 + row_l) * HID + bcol] = (__bf16)res1[r];
    }
  } else {
    // round-trip activation tile (32x128 bf16) through LDS, project via Wout
    __syncthreads();
#pragma unroll
    for (int r = 0; r < 8; ++r) {
      const int row_l = r + asel * 8;
      ldsA[row_l * HID + bcol]        = (__bf16)res0[r];
      ldsA[(row_l + 16) * HID + bcol] = (__bf16)res1[r];
    }
    __syncthreads();
    // 8 waves = 2 M-tiles x 4 N-tiles of the 32x64 output block
    const int mt = wave >> 2;
    const int nt = wave & 3;
    const int ocol = nt * 16 + arow;
    const int arow2 = mt * 16 + arow;
    v8f acc2 = {};
    const __bf16* wp = WoutPack + ((size_t)nt * 4 * 32 + lane) * 16;
    for (int ks = 0; ks < 4; ++ks) {
      const v16bf b = *reinterpret_cast<const v16bf*>(wp + (size_t)ks * 32 * 16);
      const int kb = ks * 32 + asel * 16;
      AFrag a;
      a.h[0] = *reinterpret_cast<const v8bf*>(&ldsA[arow2 * HID + kb]);
      a.h[1] = *reinterpret_cast<const v8bf*>(&ldsA[arow2 * HID + kb + 8]);
      acc2 = __builtin_amdgcn_wmma_f32_16x16x32_bf16(false, a.v, false, b, (short)0, acc2, false, false);
    }
    const float ob = bout[ocol];
#pragma unroll
    for (int r = 0; r < 8; ++r) {
      const int row_g = m0 + mt * 16 + r + asel * 8;
      fout[(size_t)row_g * OUTD + ocol] = acc2[r] + ob;
    }
  }
}

// ---------------------------------------------------------------- launcher
extern "C" void kernel_launch(void* const* d_in, const int* in_sizes, int n_in,
                              void* d_out, int out_size, void* d_ws, size_t ws_size,
                              hipStream_t stream) {
  const float* x_source = (const float*)d_in[0];
  const float* x_target = (const float*)d_in[1];
  const float* Wl_tgt   = (const float*)d_in[2];
  const float* bl_tgt   = (const float*)d_in[3];
  const float* Wr_tgt   = (const float*)d_in[4];
  const float* Wl_src   = (const float*)d_in[5];
  const float* bl_src   = (const float*)d_in[6];
  const float* Wr_src   = (const float*)d_in[7];
  const float* W_out    = (const float*)d_in[8];
  const float* b_out    = (const float*)d_in[9];
  const int* edge_src   = (const int*)d_in[10];
  const int* edge_dst   = (const int*)d_in[11];
  const int* rev_src    = (const int*)d_in[12];
  const int* rev_dst    = (const int*)d_in[13];

  char* ws = (char*)d_ws;
  size_t off = 0;
  auto alloc = [&](size_t bytes) {
    char* p = ws + off;
    off = (off + bytes + 255) & ~(size_t)255;
    return p;
  };
  float*  agg      = (float*)alloc((size_t)N_TGT * D * sizeof(float));  // 51.2 MB
  float*  cnt      = (float*)alloc((size_t)N_TGT * sizeof(float));      // contiguous after agg
  __bf16* xt1      = (__bf16*)alloc((size_t)N_TGT * D * 2);
  __bf16* xs1      = (__bf16*)alloc((size_t)N_SRC * D * 2);
  __bf16* Bpack_t0 = (__bf16*)alloc((size_t)8 * 8 * 32 * 16 * 2);  // 64 KB each
  __bf16* Bpack_s0 = (__bf16*)alloc((size_t)8 * 8 * 32 * 16 * 2);
  __bf16* Bpack_s1 = (__bf16*)alloc((size_t)8 * 8 * 32 * 16 * 2);
  __bf16* WoutPack = (__bf16*)alloc((size_t)4 * 4 * 32 * 16 * 2);  // 16 KB

  // --- weights -> bf16, pre-packed into per-lane B-fragment order
  const int WN = D * HID;  // 16384 (layer stride within Wl_*/Wr_*)
  pack_wcat_kernel<<<(8 * 8 * 32 * 16) / 256, 256, 0, stream>>>(Wl_tgt,      Wr_tgt,      Bpack_t0);
  pack_wcat_kernel<<<(8 * 8 * 32 * 16) / 256, 256, 0, stream>>>(Wl_src,      Wr_src,      Bpack_s0);
  pack_wcat_kernel<<<(8 * 8 * 32 * 16) / 256, 256, 0, stream>>>(Wl_src + WN, Wr_src + WN, Bpack_s1);
  pack_wout_kernel<<<(4 * 4 * 32 * 16) / 256, 256, 0, stream>>>(W_out, WoutPack);

  const int zn4   = (N_TGT * D + N_TGT) / 4;  // agg + cnt in one pass
  const int zgrid = (zn4 + 255) / 256;
  const int sgrid = (N_EDGES + 7) / 8;        // 1 wave per edge
  const int ggrid = N_TGT / 32;               // 3125 (exact)

  // --- layer 0, target direction: xt1 = sage(x_source -> x_target)
  zero_f4_kernel<<<zgrid, 256, 0, stream>>>((float4*)agg, zn4);
  scatter_kernel<false><<<sgrid, 256, 0, stream>>>(x_source, edge_src, edge_dst, agg, cnt, N_EDGES);
  sage_gemm_kernel<false, false><<<ggrid, 256, 0, stream>>>(
      agg, cnt, x_target, Bpack_t0, bl_tgt, xt1, nullptr, nullptr, nullptr, 0u);

  // --- layer 0, source direction: xs1 = sage(x_target -> x_source)
  zero_f4_kernel<<<zgrid, 256, 0, stream>>>((float4*)agg, zn4);
  scatter_kernel<false><<<sgrid, 256, 0, stream>>>(x_target, rev_src, rev_dst, agg, cnt, N_EDGES);
  sage_gemm_kernel<false, false><<<ggrid, 256, 0, stream>>>(
      agg, cnt, x_source, Bpack_s0, bl_src, xs1, nullptr, nullptr, nullptr, 1u);

  // --- layer 1, source direction only (layer-1 target output is dead code),
  //     with the final @W_out + b_out projection fused into the epilogue.
  zero_f4_kernel<<<zgrid, 256, 0, stream>>>((float4*)agg, zn4);
  scatter_kernel<true><<<sgrid, 256, 0, stream>>>(xt1, rev_src, rev_dst, agg, cnt, N_EDGES);
  sage_gemm_kernel<true, true><<<ggrid, 256, 0, stream>>>(
      agg, cnt, xs1, Bpack_s1, bl_src + HID, nullptr, WoutPack, b_out, (float*)d_out, 3u);
}
